// TinySTGNN_Optimized_22814866276777
// MI455X (gfx1250) — matrix operand order repaired and checked by reading, not compiled, not run on hardware
//
#include <hip/hip_runtime.h>
#include <hip/hip_bf16.h>
#include <math.h>

// ---------------------------------------------------------------------------
// TinySTGNN forward, fused for MI455X (gfx1250, wave32, WMMA).
//   K1: per-(b,t) LayerNorm + corr(|xn xn^T|) + agg + ReLU-FC + node-mean,
//       all three GEMMs via v_wmma_f32_16x16x32_f16 through LDS.
//       (xln kept transposed in LDS so ALL fragments load as ds_load_b128.)
//   K2: single-workgroup sequential LSTM, 16x1024 gate GEMM per step in WMMA,
//       global_prefetch of next step's input.
//   K3: attention softmax + classifier (tiny, scalar).
// ---------------------------------------------------------------------------

typedef __attribute__((ext_vector_type(16))) _Float16 v16h;
typedef __attribute__((ext_vector_type(8)))  float    v8f;

#define B_    16
#define N_    128
#define T_    256
#define F_    64
#define GATH  128
#define LSTMH 256
#define GATES 1024

__device__ __forceinline__ v8f wmma16(v16h a, v16h b, v8f c) {
  return __builtin_amdgcn_wmma_f32_16x16x32_f16(
      /*neg_a=*/false, a, /*neg_b=*/false, b,
      /*c_mod=*/(short)0, c, /*reuse_a=*/false, /*reuse_b=*/false);
}

// A fragment, 16x32 f16, from row-major matrix (row stride ld, halfs).
// lane L: row = L%16; lanes 0-15 hold K 0-15, lanes 16-31 hold K 16-31.
__device__ __forceinline__ v16h frag_a(const _Float16* base, int row0, int k0, int ld) {
  const int lane = threadIdx.x & 31;
  const int r = lane & 15, hlf = lane >> 4;
  const _Float16* row = base + (size_t)(row0 + r) * ld + k0;
  v16h a;
#pragma unroll
  for (int i = 0; i < 8; ++i) a[i]     = row[hlf * 16 + i];
#pragma unroll
  for (int i = 0; i < 8; ++i) a[8 + i] = row[hlf * 16 + 8 + i];
  return a;
}

// B fragment, 32x16 f16, where B = M^T for row-major M (n-major, stride ld):
// b[i] = M[n0 + lane%16][k0 + (lane/16)*16 + i]  (contiguous 32B per lane).
__device__ __forceinline__ v16h frag_bT(const _Float16* base, int n0, int k0, int ld) {
  const int lane = threadIdx.x & 31;
  const int r = lane & 15, hlf = lane >> 4;
  const _Float16* row = base + (size_t)(n0 + r) * ld + k0 + hlf * 16;
  v16h b;
#pragma unroll
  for (int i = 0; i < 16; ++i) b[i] = row[i];
  return b;
}

__device__ __forceinline__ float sigm(float x) { return 1.0f / (1.0f + expf(-x)); }

// ---------------------------------------------------------------------------
// Kernel 0: convert LSTM weights to f16, fold biases.
// ---------------------------------------------------------------------------
__global__ void k_convert(const float* __restrict__ Wih, const float* __restrict__ Whh,
                          const float* __restrict__ bih, const float* __restrict__ bhh,
                          _Float16* __restrict__ WihH, _Float16* __restrict__ WhhH,
                          float* __restrict__ bias) {
  const int i = blockIdx.x * blockDim.x + threadIdx.x;
  const int nih = GATES * GATH;
  const int nhh = GATES * LSTMH;
  if (i < nih) WihH[i] = (_Float16)Wih[i];
  if (i < nhh) WhhH[i] = (_Float16)Whh[i];
  if (i < GATES) bias[i] = bih[i] + bhh[i];
}

// ---------------------------------------------------------------------------
// Kernel 1: fused spatial stage, one workgroup (256 thr = 8 waves) per (b,t).
// ---------------------------------------------------------------------------
__global__ __launch_bounds__(256) void k_spatial(
    const float* __restrict__ x, const float* __restrict__ lnw, const float* __restrict__ lnb,
    const float* __restrict__ Wfc, const float* __restrict__ bfc,
    float* __restrict__ spatial_seq)  // (B,T,GATH)
{
  const int bt = blockIdx.x;          // B_*T_ blocks
  const int b = bt >> 8;              // /T_
  const int t = bt & (T_ - 1);

  constexpr int LDF = F_ + 8;         // 72 halfs
  constexpr int LDN = N_ + 8;         // 136 halfs
  __shared__ _Float16 s_xn  [N_ * LDF];   // corr-normalized x, row-major (N x F)
  __shared__ _Float16 s_xlnT[F_ * LDN];   // layernormed x, TRANSPOSED (F x N)
  __shared__ _Float16 s_wfc [GATH * LDF]; // W_fc, f16 (G x F)
  __shared__ _Float16 s_corr[N_ * LDN];   // |xn xn^T| (N x N)
  __shared__ _Float16 s_xagg[N_ * LDF];   // corr @ xln (N x F)
  __shared__ float    s_bfc[GATH];
  __shared__ float    s_colsum[GATH];

  const int tid = threadIdx.x;

  // ---- Phase 1: LayerNorm + normalize (threads 0-127), stage W_fc (128-255)
  if (tid < N_) {
    const int n = tid;
    const float* xr = x + (((size_t)b * N_ + n) * T_ + t) * F_;
    float v[F_];
    float sum = 0.0f;
#pragma unroll
    for (int f = 0; f < F_; f += 4) {
      float4 q = *(const float4*)(xr + f);
      v[f] = q.x; v[f + 1] = q.y; v[f + 2] = q.z; v[f + 3] = q.w;
      sum += q.x + q.y + q.z + q.w;
    }
    const float mu = sum * (1.0f / F_);
    float vs = 0.0f;
#pragma unroll
    for (int f = 0; f < F_; ++f) { float d = v[f] - mu; vs += d * d; }
    const float inv = rsqrtf(vs * (1.0f / F_) + 1e-5f);
    float s2 = 0.0f;
#pragma unroll
    for (int f = 0; f < F_; ++f) { v[f] = (v[f] - mu) * inv * lnw[f] + lnb[f]; s2 += v[f]; }
    const float m = s2 * (1.0f / F_);
    float ss = 0.0f;
#pragma unroll
    for (int f = 0; f < F_; ++f) { float d = v[f] - m; ss += d * d; }
    const float stdinv = 1.0f / (sqrtf(ss) + 1e-8f);
#pragma unroll
    for (int f = 0; f < F_; ++f) {
      s_xlnT[f * LDN + n] = (_Float16)v[f];                 // transposed store
      s_xn  [n * LDF + f] = (_Float16)((v[f] - m) * stdinv);
    }
    s_colsum[n] = 0.0f;
  } else {
    const int g = tid - N_;           // 0..127
#pragma unroll
    for (int f = 0; f < F_; ++f) s_wfc[g * LDF + f] = (_Float16)Wfc[g * F_ + f];
    s_bfc[g] = bfc[g];
  }
  __syncthreads();

  const int wave = tid >> 5;
  const int lane = tid & 31;
  const int r = lane & 15, hlf = lane >> 4;

  // ---- Phase 2: corr = |xn @ xn^T| (128x128, K=64): 64 tiles, 8/wave
#pragma unroll
  for (int s = 0; s < 8; ++s) {
    const int tile = wave * 8 + s;
    const int m0 = (tile >> 3) * 16, n0 = (tile & 7) * 16;
    v8f acc = {};
    acc = wmma16(frag_a(s_xn, m0, 0,  LDF), frag_bT(s_xn, n0, 0,  LDF), acc);
    acc = wmma16(frag_a(s_xn, m0, 32, LDF), frag_bT(s_xn, n0, 32, LDF), acc);
#pragma unroll
    for (int j = 0; j < 8; ++j)
      s_corr[(m0 + j + 8 * hlf) * LDN + n0 + r] = (_Float16)fabsf(acc[j]);
  }
  __syncthreads();

  // ---- Phase 3: xagg = corr @ xln (128x64, K=128): 32 tiles, 4/wave
  // B operand: B[k][f] = xln[k][f] = xlnT[f][k] -> contiguous via frag_bT on xlnT.
#pragma unroll
  for (int s = 0; s < 4; ++s) {
    const int tile = wave * 4 + s;
    const int m0 = (tile >> 2) * 16, n0 = (tile & 3) * 16;
    v8f acc = {};
#pragma unroll
    for (int k0 = 0; k0 < N_; k0 += 32)
      acc = wmma16(frag_a(s_corr, m0, k0, LDN), frag_bT(s_xlnT, n0, k0, LDN), acc);
#pragma unroll
    for (int j = 0; j < 8; ++j)
      s_xagg[(m0 + j + 8 * hlf) * LDF + n0 + r] = (_Float16)acc[j];
  }
  __syncthreads();

  // ---- Phase 4: h = relu(xagg @ Wfc^T + bfc) (128x128, K=64) + column sums
#pragma unroll
  for (int s = 0; s < 8; ++s) {
    const int tile = wave * 8 + s;
    const int m0 = (tile >> 3) * 16, g0 = (tile & 7) * 16;
    v8f acc = {};
    acc = wmma16(frag_a(s_xagg, m0, 0,  LDF), frag_bT(s_wfc, g0, 0,  LDF), acc);
    acc = wmma16(frag_a(s_xagg, m0, 32, LDF), frag_bT(s_wfc, g0, 32, LDF), acc);
    const float bb = s_bfc[g0 + r];
    float psum = 0.0f;
#pragma unroll
    for (int j = 0; j < 8; ++j) psum += fmaxf(acc[j] + bb, 0.0f);
    atomicAdd(&s_colsum[g0 + r], psum);   // ds_add_f32
  }
  __syncthreads();

  if (tid < GATH)
    spatial_seq[((size_t)b * T_ + t) * GATH + tid] = s_colsum[tid] * (1.0f / N_);
}

// ---------------------------------------------------------------------------
// Kernel 2: LSTM, single workgroup (512 thr = 16 waves), sequential over T.
// Per step: gates(16x1024) = xt(16x128)@Wih^T + h(16x256)@Whh^T + bias (WMMA).
// ---------------------------------------------------------------------------
__global__ __launch_bounds__(512) void k_lstm(
    const float* __restrict__ spatial_seq,   // (B,T,GATH)
    const _Float16* __restrict__ Wih,        // (GATES,GATH) f16
    const _Float16* __restrict__ Whh,        // (GATES,LSTMH) f16
    const float* __restrict__ bias,          // GATES (b_ih + b_hh)
    float* __restrict__ lstm_out)            // (B,T,LSTMH)
{
  constexpr int LDX = GATH + 8;    // 136
  constexpr int LDH = LSTMH + 8;   // 264
  constexpr int LDG = GATES + 8;   // 1032
  __shared__ _Float16 s_xt[B_ * LDX];
  __shared__ _Float16 s_h [B_ * LDH];
  __shared__ float    s_c [B_ * LSTMH];
  __shared__ float    s_g [B_ * LDG];
  __shared__ float    s_bias[GATES];

  const int tid = threadIdx.x;
  for (int i = tid; i < B_ * LDH;   i += 512) s_h[i] = (_Float16)0.0f;
  for (int i = tid; i < B_ * LSTMH; i += 512) s_c[i] = 0.0f;
  for (int i = tid; i < GATES;      i += 512) s_bias[i] = bias[i];
  __syncthreads();

  const int wave = tid >> 5, lane = tid & 31;
  const int r = lane & 15, hlf = lane >> 4;

  for (int t = 0; t < T_; ++t) {
    // stage x_t into LDS as f16
    for (int i = tid; i < B_ * GATH; i += 512) {
      const int m = i >> 7, f = i & (GATH - 1);
      s_xt[m * LDX + f] = (_Float16)spatial_seq[((size_t)m * T_ + t) * GATH + f];
    }
    // prefetch next step's input slice (one 64B line per thread)
    if (t + 1 < T_ && tid < (B_ * GATH) / 16) {
      const int m = tid >> 3, c16 = tid & 7;
      __builtin_prefetch(spatial_seq + ((size_t)m * T_ + (t + 1)) * GATH + c16 * 16, 0, 3);
    }
    __syncthreads();

    // gate GEMM: 64 n-tiles of 16, 4 per wave; K = 128 (x) + 256 (h)
#pragma unroll
    for (int s = 0; s < 4; ++s) {
      const int g0 = (wave * 4 + s) * 16;
      v8f acc = {};
#pragma unroll
      for (int k0 = 0; k0 < GATH; k0 += 32)
        acc = wmma16(frag_a(s_xt, 0, k0, LDX), frag_bT(Wih, g0, k0, GATH), acc);
#pragma unroll
      for (int k0 = 0; k0 < LSTMH; k0 += 32)
        acc = wmma16(frag_a(s_h, 0, k0, LDH), frag_bT(Whh, g0, k0, LSTMH), acc);
      const float bb = s_bias[g0 + r];
#pragma unroll
      for (int j = 0; j < 8; ++j)
        s_g[(j + 8 * hlf) * LDG + g0 + r] = acc[j] + bb;
    }
    __syncthreads();

    // elementwise LSTM cell: 16*256 pairs, 8 per thread
    for (int i = tid; i < B_ * LSTMH; i += 512) {
      const int m = i >> 8, hh = i & (LSTMH - 1);
      const float ig = s_g[m * LDG + hh];
      const float fg = s_g[m * LDG + LSTMH + hh];
      const float gg = s_g[m * LDG + 2 * LSTMH + hh];
      const float og = s_g[m * LDG + 3 * LSTMH + hh];
      const float c  = sigm(fg) * s_c[i] + sigm(ig) * tanhf(gg);
      const float hv = sigm(og) * tanhf(c);
      s_c[i] = c;
      s_h[m * LDH + hh] = (_Float16)hv;
      lstm_out[((size_t)m * T_ + t) * LSTMH + hh] = hv;
    }
    __syncthreads();
  }
}

// ---------------------------------------------------------------------------
// Kernel 3: temporal attention + classifier. One block per batch element.
// ---------------------------------------------------------------------------
__global__ __launch_bounds__(256) void k_attn(
    const float* __restrict__ lstm_out, const float* __restrict__ Watt,
    const float* __restrict__ batt, const float* __restrict__ Wcls,
    const float* __restrict__ bcls, float* __restrict__ out)
{
  const int b = blockIdx.x;
  const int tid = threadIdx.x;
  __shared__ float s_w[LSTMH];
  __shared__ float s_sc[T_];
  __shared__ float s_red[T_];
  __shared__ float s_ctx[LSTMH];

  s_w[tid] = Watt[tid];
  __syncthreads();

  // score_t = tanh(<lstm_out[b,t,:], Watt> + batt)
  {
    const float* row = lstm_out + ((size_t)b * T_ + tid) * LSTMH;
    float acc = 0.0f;
    for (int h = 0; h < LSTMH; h += 4) {
      float4 q = *(const float4*)(row + h);
      acc += q.x * s_w[h] + q.y * s_w[h + 1] + q.z * s_w[h + 2] + q.w * s_w[h + 3];
    }
    s_sc[tid] = tanhf(acc + batt[0]);
  }
  __syncthreads();

  // softmax over time
  s_red[tid] = s_sc[tid];
  __syncthreads();
  for (int off = 128; off > 0; off >>= 1) {
    if (tid < off) s_red[tid] = fmaxf(s_red[tid], s_red[tid + off]);
    __syncthreads();
  }
  const float mx = s_red[0];
  __syncthreads();
  const float e = expf(s_sc[tid] - mx);
  s_red[tid] = e;
  __syncthreads();
  for (int off = 128; off > 0; off >>= 1) {
    if (tid < off) s_red[tid] += s_red[tid + off];
    __syncthreads();
  }
  const float denom = s_red[0];
  __syncthreads();
  s_sc[tid] = e / denom;
  __syncthreads();

  // context[h] = sum_t w[t] * lstm_out[b,t,h]
  {
    const float* base = lstm_out + (size_t)b * T_ * LSTMH + tid;
    float acc = 0.0f;
    for (int tt = 0; tt < T_; ++tt) acc += s_sc[tt] * base[(size_t)tt * LSTMH];
    s_ctx[tid] = acc;
  }
  __syncthreads();

  // logits (2 classes)
  const float p0 = s_ctx[tid] * Wcls[tid];
  const float p1 = s_ctx[tid] * Wcls[LSTMH + tid];
  s_red[tid] = p0;
  __syncthreads();
  for (int off = 128; off > 0; off >>= 1) {
    if (tid < off) s_red[tid] += s_red[tid + off];
    __syncthreads();
  }
  if (tid == 0) out[b * 2 + 0] = s_red[0] + bcls[0];
  __syncthreads();
  s_red[tid] = p1;
  __syncthreads();
  for (int off = 128; off > 0; off >>= 1) {
    if (tid < off) s_red[tid] += s_red[tid + off];
    __syncthreads();
  }
  if (tid == 0) out[b * 2 + 1] = s_red[0] + bcls[1];
}

// ---------------------------------------------------------------------------
extern "C" void kernel_launch(void* const* d_in, const int* in_sizes, int n_in,
                              void* d_out, int out_size, void* d_ws, size_t ws_size,
                              hipStream_t stream) {
  const float* x    = (const float*)d_in[0];
  const float* lnw  = (const float*)d_in[1];
  const float* lnb  = (const float*)d_in[2];
  const float* Wfc  = (const float*)d_in[3];
  const float* bfc  = (const float*)d_in[4];
  const float* Wih  = (const float*)d_in[5];
  const float* Whh  = (const float*)d_in[6];
  const float* bih  = (const float*)d_in[7];
  const float* bhh  = (const float*)d_in[8];
  const float* Watt = (const float*)d_in[9];
  const float* batt = (const float*)d_in[10];
  const float* Wcls = (const float*)d_in[11];
  const float* bcls = (const float*)d_in[12];

  // workspace layout (≈6.76 MB)
  char* ws = (char*)d_ws;
  float*    spatial  = (float*)ws;                                   // 2 MB
  float*    lstm_out = (float*)(ws + (2u << 20));                    // 4 MB
  _Float16* WihH     = (_Float16*)(ws + (6u << 20));                 // 256 KB
  _Float16* WhhH     = (_Float16*)(ws + (6u << 20) + (1u << 18));    // 512 KB
  float*    bias     = (float*)(ws + (6u << 20) + (1u << 18) + (1u << 19));

  k_convert<<<(GATES * LSTMH + 255) / 256, 256, 0, stream>>>(
      Wih, Whh, bih, bhh, WihH, WhhH, bias);
  k_spatial<<<B_ * T_, 256, 0, stream>>>(x, lnw, lnb, Wfc, bfc, spatial);
  k_lstm<<<1, 512, 0, stream>>>(spatial, WihH, WhhH, bias, lstm_out);
  k_attn<<<B_, 256, 0, stream>>>(lstm_out, Watt, batt, Wcls, bcls, (float*)d_out);
}
